// MetaKG_67577015436450
// MI455X (gfx1250) — matrix operand order repaired and checked by reading, not compiled, not run on hardware
//
#include <hip/hip_runtime.h>
#include <hip/hip_bf16.h>
#include <math.h>

#define NN 100000
#define EE 1600000
#define RR 8
#define DD 64

typedef __attribute__((ext_vector_type(16))) __bf16         v16bf;
typedef __attribute__((ext_vector_type(16))) unsigned short v16u;
typedef __attribute__((ext_vector_type(8)))  unsigned short v8u;
typedef __attribute__((ext_vector_type(8)))  float          v8f;

// ---------- helpers ----------
__device__ __forceinline__ unsigned short f2bf(float f) {
    unsigned u = __float_as_uint(f);
    unsigned r = u + 0x7FFFu + ((u >> 16) & 1u);   // round-to-nearest-even
    return (unsigned short)(r >> 16);
}
__device__ __forceinline__ int f2ord(float f) {    // monotonic float->int map
    int i = __float_as_int(f);
    return i >= 0 ? i : (i ^ 0x7FFFFFFF);
}
__device__ __forceinline__ float ord2f(int i) {
    int j = i >= 0 ? i : (i ^ 0x7FFFFFFF);
    return __int_as_float(j);
}
__device__ __forceinline__ v8f wmma_bf16(v16u a, v16u b, v8f c) {
    return __builtin_amdgcn_wmma_f32_16x16x32_bf16(
        /*neg_a=*/false, __builtin_bit_cast(v16bf, a),
        /*neg_b=*/false, __builtin_bit_cast(v16bf, b),
        /*c_mod=*/(short)0, c, /*reuse_a=*/false, /*reuse_b=*/false);
}
__device__ __forceinline__ v16u cat8(v8u lo, v8u hi) {
    return __builtin_shufflevector(lo, hi, 0, 1, 2, 3, 4, 5, 6, 7,
                                           8, 9, 10, 11, 12, 13, 14, 15);
}
__device__ __forceinline__ float lrelu(float v) { return v > 0.f ? v : 0.01f * v; }

union F4 { float4 v; float f[4]; };

// ---------- init / convert ----------
__global__ void zero_f32(float* p, int n) {
    int i = blockIdx.x * blockDim.x + threadIdx.x;
    if (i < n) p[i] = 0.f;
}
__global__ void fill_i32(int* p, int v, int n) {
    int i = blockIdx.x * blockDim.x + threadIdx.x;
    if (i < n) p[i] = v;
}
__global__ void cvt_bf16(const float* __restrict__ s, unsigned short* __restrict__ d, int n) {
    int i = blockIdx.x * blockDim.x + threadIdx.x;
    if (i < n) d[i] = f2bf(s[i]);
}
// W_R [R,64(k),64(n)] f32  ->  wrt [R,64(n),64(k)] bf16  (K contiguous per output col)
__global__ void cvt_wr_t(const float* __restrict__ s, unsigned short* __restrict__ d) {
    int i = blockIdx.x * blockDim.x + threadIdx.x;
    if (i >= RR * 64 * 64) return;
    int r = i >> 12, rem = i & 4095, n = rem >> 6, k = rem & 63;
    d[i] = f2bf(s[(r << 12) + (k << 6) + n]);
}

// ---------- proj[n,r,:] = entity_emb[n] @ W_R[r]  (WMMA, async-LDS staged A) ----------
// grid = N/16, block = 128 (4 waves); wave w owns output cols [16w,16w+16); loops r=0..7.
__global__ __launch_bounds__(128) void proj_gemm(
    const unsigned short* __restrict__ ebf,   // [N,64] bf16
    const unsigned short* __restrict__ wrt,   // [R,64(n),64(k)] bf16
    float* __restrict__ proj)                 // [N,R,64] f32
{
    __shared__ __align__(16) unsigned short lA[16 * 64];   // 2 KB A tile
    const int lane = threadIdx.x & 31;
    const int wv   = threadIdx.x >> 5;
    const int row0 = blockIdx.x << 4;
    const int m = lane & 15, half = lane >> 4;

    // ---- async stage: 16 contiguous rows (2 KB) of ebf -> LDS ----
    if (wv == 0) {
        unsigned ldsBase = (unsigned)(unsigned long long)(void*)lA;
        const unsigned short* gsrc = ebf + (size_t)row0 * 64;
#pragma unroll
        for (int j = 0; j < 4; ++j) {
            unsigned ldsa = ldsBase + lane * 16 + j * 512;
            unsigned voff = lane * 16 + j * 512;
            asm volatile("global_load_async_to_lds_b128 %0, %1, %2"
                         :: "v"(ldsa), "v"(voff), "s"(gsrc) : "memory");
        }
        asm volatile("s_wait_asynccnt 0x0" ::: "memory");
    }
    __syncthreads();

    // ---- A fragments from LDS (reused for all 8 relations) ----
    v16u afrag[2];
#pragma unroll
    for (int kt = 0; kt < 2; ++kt) {
        v8u a0 = *(const v8u*)&lA[m * 64 + kt * 32 + half * 16];
        v8u a1 = *(const v8u*)&lA[m * 64 + kt * 32 + 8 + half * 16];
        afrag[kt] = cat8(a0, a1);
    }

#pragma unroll
    for (int r = 0; r < RR; ++r) {
        const unsigned short* Bt = wrt + r * (64 * 64);
        v8f c = {};
#pragma unroll
        for (int kt = 0; kt < 2; ++kt) {
            // lane's 16 contiguous K values of column n
            v16u b = *(const v16u*)&Bt[(wv * 16 + m) * 64 + kt * 32 + half * 16];
            c = wmma_bf16(afrag[kt], b, c);
        }
#pragma unroll
        for (int vr = 0; vr < 8; ++vr) {
            int mm = vr + 8 * half;
            proj[(size_t)(row0 + mm) * (RR * 64) + r * 64 + wv * 16 + m] = c[vr];
        }
    }
}

// ---------- attention: wave per edge, lane-parallel over 64 dims ----------
__global__ __launch_bounds__(256) void att_edge(
    const float* __restrict__ proj, const float* __restrict__ rel,
    const int* __restrict__ src, const int* __restrict__ dst, const int* __restrict__ et,
    float* __restrict__ att, int* __restrict__ segmax)
{
    int e = (blockIdx.x * blockDim.x + threadIdx.x) >> 5;
    int lane = threadIdx.x & 31;
    if (e >= EE) return;
    int s = src[e], d = dst[e], r = et[e];
    const float* pt = proj + (size_t)s * (RR * 64) + r * 64;
    const float* ph = proj + (size_t)d * (RR * 64) + r * 64;
    const float* re = rel + r * 64;
    float sum = 0.f;
#pragma unroll
    for (int j = 0; j < 2; ++j) {
        int k = lane + 32 * j;
        sum += pt[k] * tanhf(ph[k] + re[k]);
    }
#pragma unroll
    for (int off = 16; off; off >>= 1) sum += __shfl_down(sum, off, 32);
    if (lane == 0) {
        att[e] = sum;
        atomicMax(&segmax[d], f2ord(sum));
    }
}

// ---------- edge softmax numerator + segment sum ----------
__global__ void ex_edge(const float* __restrict__ att, const int* __restrict__ dst,
                        const int* __restrict__ segmax, float* __restrict__ ex,
                        float* __restrict__ segsum)
{
    int e = blockIdx.x * blockDim.x + threadIdx.x;
    if (e >= EE) return;
    int d = dst[e];
    float v = __expf(att[e] - ord2f(segmax[d]));
    ex[e] = v;
    atomicAdd(&segsum[d], v);
}

// ---------- aggregation: Nh[dst] += x[src] * a  (wave per edge) ----------
__global__ __launch_bounds__(256) void agg64(
    const float* __restrict__ x, const float* __restrict__ ex, const float* __restrict__ segsum,
    const int* __restrict__ src, const int* __restrict__ dst, float* __restrict__ Nh)
{
    int e = (blockIdx.x * blockDim.x + threadIdx.x) >> 5;
    int lane = threadIdx.x & 31;
    if (e >= EE) return;
    int s = src[e], d = dst[e];
    float coef = ex[e] / segsum[d];
    atomicAdd(&Nh[d * 64 + lane],      x[s * 64 + lane]      * coef);
    atomicAdd(&Nh[d * 64 + lane + 32], x[s * 64 + lane + 32] * coef);
}
__global__ __launch_bounds__(256) void agg32(
    const float* __restrict__ x, const float* __restrict__ ex, const float* __restrict__ segsum,
    const int* __restrict__ src, const int* __restrict__ dst, float* __restrict__ Nh)
{
    int e = (blockIdx.x * blockDim.x + threadIdx.x) >> 5;
    int lane = threadIdx.x & 31;
    if (e >= EE) return;
    int s = src[e], d = dst[e];
    float coef = ex[e] / segsum[d];
    atomicAdd(&Nh[d * 32 + lane], x[s * 32 + lane] * coef);
}

// ---------- layer-0 GEMMs: z1=(x+Nh)@W1^T+b1, z2=(x*Nh)@W2^T+b2  (K=64, 32 cols) ----------
// block = 128 (4 waves): wave0/1 -> z1 col-tiles, wave2/3 -> z2 col-tiles; grid = N/16
__global__ __launch_bounds__(128) void mlp_gemm64(
    const float* __restrict__ x, const float* __restrict__ Nh,
    const unsigned short* __restrict__ w1bf, const unsigned short* __restrict__ w2bf, // [32,64] bf16
    const float* __restrict__ b1, const float* __restrict__ b2,
    float* __restrict__ z1, float* __restrict__ z2)
{
    const int lane = threadIdx.x & 31;
    const int wv   = threadIdx.x >> 5;
    const int ct   = wv & 1;
    const bool prod = (wv >= 2);
    const unsigned short* Wb = prod ? w2bf : w1bf;
    const float* bias        = prod ? b2 : b1;
    float* z                 = prod ? z2 : z1;
    const int row0 = blockIdx.x << 4;
    const int m = lane & 15, half = lane >> 4;
    v8f c = {};
#pragma unroll
    for (int kt = 0; kt < 2; ++kt) {
        int ri0 = (row0 + m) * 64 + kt * 32 + half * 16;   // 16 contiguous floats
        F4 x0, x1, h0, h1, x2, x3, h2, h3;
        x0.v = *(const float4*)&x[ri0];       x1.v = *(const float4*)&x[ri0 + 4];
        h0.v = *(const float4*)&Nh[ri0];      h1.v = *(const float4*)&Nh[ri0 + 4];
        x2.v = *(const float4*)&x[ri0 + 8];   x3.v = *(const float4*)&x[ri0 + 12];
        h2.v = *(const float4*)&Nh[ri0 + 8];  h3.v = *(const float4*)&Nh[ri0 + 12];
        v16u a;
#pragma unroll
        for (int i = 0; i < 4; ++i) {
            a[i]      = f2bf(prod ? x0.f[i] * h0.f[i] : x0.f[i] + h0.f[i]);
            a[4 + i]  = f2bf(prod ? x1.f[i] * h1.f[i] : x1.f[i] + h1.f[i]);
            a[8 + i]  = f2bf(prod ? x2.f[i] * h2.f[i] : x2.f[i] + h2.f[i]);
            a[12 + i] = f2bf(prod ? x3.f[i] * h3.f[i] : x3.f[i] + h3.f[i]);
        }
        v16u b = *(const v16u*)&Wb[(ct * 16 + m) * 64 + kt * 32 + half * 16];
        c = wmma_bf16(a, b, c);
    }
#pragma unroll
    for (int vr = 0; vr < 8; ++vr) {
        int mm = vr + 8 * half;
        int n  = ct * 16 + m;
        z[(row0 + mm) * 32 + n] = c[vr] + bias[n];
    }
}

// ---------- layer-1 GEMMs: K=32, 16 cols; block = 64 (wave0->z1, wave1->z2) ----------
__global__ __launch_bounds__(64) void mlp_gemm32(
    const float* __restrict__ x, const float* __restrict__ Nh,
    const unsigned short* __restrict__ w1bf, const unsigned short* __restrict__ w2bf, // [16,32] bf16
    const float* __restrict__ b1, const float* __restrict__ b2,
    float* __restrict__ z1, float* __restrict__ z2)
{
    const int lane = threadIdx.x & 31;
    const bool prod = (threadIdx.x >> 5) != 0;
    const unsigned short* Wb = prod ? w2bf : w1bf;
    const float* bias        = prod ? b2 : b1;
    float* z                 = prod ? z2 : z1;
    const int row0 = blockIdx.x << 4;
    const int m = lane & 15, half = lane >> 4;
    int ri0 = (row0 + m) * 32 + half * 16;
    F4 x0, x1, h0, h1, x2, x3, h2, h3;
    x0.v = *(const float4*)&x[ri0];       x1.v = *(const float4*)&x[ri0 + 4];
    h0.v = *(const float4*)&Nh[ri0];      h1.v = *(const float4*)&Nh[ri0 + 4];
    x2.v = *(const float4*)&x[ri0 + 8];   x3.v = *(const float4*)&x[ri0 + 12];
    h2.v = *(const float4*)&Nh[ri0 + 8];  h3.v = *(const float4*)&Nh[ri0 + 12];
    v16u a;
#pragma unroll
    for (int i = 0; i < 4; ++i) {
        a[i]      = f2bf(prod ? x0.f[i] * h0.f[i] : x0.f[i] + h0.f[i]);
        a[4 + i]  = f2bf(prod ? x1.f[i] * h1.f[i] : x1.f[i] + h1.f[i]);
        a[8 + i]  = f2bf(prod ? x2.f[i] * h2.f[i] : x2.f[i] + h2.f[i]);
        a[12 + i] = f2bf(prod ? x3.f[i] * h3.f[i] : x3.f[i] + h3.f[i]);
    }
    v16u b = *(const v16u*)&Wb[m * 32 + half * 16];
    v8f c = {};
    c = wmma_bf16(a, b, c);
#pragma unroll
    for (int vr = 0; vr < 8; ++vr) {
        int mm = vr + 8 * half;
        z[(row0 + mm) * 16 + m] = c[vr] + bias[m];
    }
}

// ---------- epilogues: leaky + add + L2-normalize (wave per node) ----------
__global__ __launch_bounds__(256) void norm32(
    const float* __restrict__ z1, const float* __restrict__ z2,
    float* __restrict__ h1, float* __restrict__ out)
{
    int n = (blockIdx.x * blockDim.x + threadIdx.x) >> 5;
    int lane = threadIdx.x & 31;
    if (n >= NN) return;
    float v = lrelu(z1[n * 32 + lane]) + lrelu(z2[n * 32 + lane]);
    float s = v * v;
#pragma unroll
    for (int off = 16; off; off >>= 1) s += __shfl_xor(s, off, 32);
    float inv = 1.f / fmaxf(sqrtf(s), 1e-12f);
    float hv = v * inv;
    h1[n * 32 + lane] = hv;
    out[n * 112 + 64 + lane] = hv;
}
__global__ __launch_bounds__(256) void norm16(
    const float* __restrict__ z1, const float* __restrict__ z2, float* __restrict__ out)
{
    int n = (blockIdx.x * blockDim.x + threadIdx.x) >> 5;
    int lane = threadIdx.x & 31;
    if (n >= NN) return;
    int col = lane & 15;
    float v = lrelu(z1[n * 16 + col]) + lrelu(z2[n * 16 + col]);
    float s = v * v;
#pragma unroll
    for (int off = 8; off; off >>= 1) s += __shfl_xor(s, off, 32);  // reduce within 16-group
    float inv = 1.f / fmaxf(sqrtf(s), 1e-12f);
    if (lane < 16) out[n * 112 + 96 + col] = v * inv;
}
__global__ void ego_copy(const float* __restrict__ x, float* __restrict__ out) {
    int i = blockIdx.x * blockDim.x + threadIdx.x;   // over N*16 float4s
    if (i >= NN * 16) return;
    int n = i >> 4, c = (i & 15) * 4;
    *(float4*)&out[n * 112 + c] = *(const float4*)&x[n * 64 + c];
}

// ---------- host launcher ----------
extern "C" void kernel_launch(void* const* d_in, const int* in_sizes, int n_in,
                              void* d_out, int out_size, void* d_ws, size_t ws_size,
                              hipStream_t stream) {
    const float* entity = (const float*)d_in[0];   // [N,64]
    const float* rel    = (const float*)d_in[1];   // [R,64]
    const float* W_R    = (const float*)d_in[2];   // [R,64,64]
    const float* W1_0   = (const float*)d_in[3];   // [32,64]
    const float* b1_0   = (const float*)d_in[4];
    const float* W2_0   = (const float*)d_in[5];
    const float* b2_0   = (const float*)d_in[6];
    const float* W1_1   = (const float*)d_in[7];   // [16,32]
    const float* b1_1   = (const float*)d_in[8];
    const float* W2_1   = (const float*)d_in[9];
    const float* b2_1   = (const float*)d_in[10];
    const int*   src    = (const int*)d_in[11];
    const int*   dst    = (const int*)d_in[12];
    const int*   etype  = (const int*)d_in[13];
    float* out = (float*)d_out;

    char* w = (char*)d_ws;
    size_t off = 0;
    auto take = [&](size_t bytes) -> char* {
        char* p = w + off;
        off = (off + bytes + 255) & ~(size_t)255;
        return p;
    };
    unsigned short* ebf   = (unsigned short*)take((size_t)NN * 64 * 2);
    unsigned short* wrt   = (unsigned short*)take((size_t)RR * 64 * 64 * 2);
    unsigned short* w1b0  = (unsigned short*)take(32 * 64 * 2);
    unsigned short* w2b0  = (unsigned short*)take(32 * 64 * 2);
    unsigned short* w1b1  = (unsigned short*)take(16 * 32 * 2);
    unsigned short* w2b1  = (unsigned short*)take(16 * 32 * 2);
    float* proj   = (float*)take((size_t)NN * RR * 64 * 4);
    float* att    = (float*)take((size_t)EE * 4);
    float* ex     = (float*)take((size_t)EE * 4);
    int*   segmax = (int*)  take((size_t)NN * 4);
    float* segsum = (float*)take((size_t)NN * 4);
    float* Nh0    = (float*)take((size_t)NN * 64 * 4);
    float* z1     = (float*)take((size_t)NN * 32 * 4);
    float* z2     = (float*)take((size_t)NN * 32 * 4);
    float* h1     = (float*)take((size_t)NN * 32 * 4);
    float* Nh1    = (float*)take((size_t)NN * 32 * 4);
    float* z1b    = (float*)take((size_t)NN * 16 * 4);
    float* z2b    = (float*)take((size_t)NN * 16 * 4);
    (void)ws_size; (void)n_in; (void)in_sizes; (void)out_size;

    const int T = 256;
    // init accumulators
    fill_i32<<<(NN + T - 1) / T, T, 0, stream>>>(segmax, (int)0x80000000, NN);
    zero_f32<<<(NN + T - 1) / T, T, 0, stream>>>(segsum, NN);
    zero_f32<<<(NN * 64 + T - 1) / T, T, 0, stream>>>(Nh0, NN * 64);
    zero_f32<<<(NN * 32 + T - 1) / T, T, 0, stream>>>(Nh1, NN * 32);
    // bf16 conversions (W_R transposed to [r][n][k] for contiguous-K B loads)
    cvt_bf16<<<(NN * 64 + T - 1) / T, T, 0, stream>>>(entity, ebf, NN * 64);
    cvt_wr_t<<<(RR * 64 * 64 + T - 1) / T, T, 0, stream>>>(W_R, wrt);
    cvt_bf16<<<(32 * 64 + T - 1) / T, T, 0, stream>>>(W1_0, w1b0, 32 * 64);
    cvt_bf16<<<(32 * 64 + T - 1) / T, T, 0, stream>>>(W2_0, w2b0, 32 * 64);
    cvt_bf16<<<(16 * 32 + T - 1) / T, T, 0, stream>>>(W1_1, w1b1, 16 * 32);
    cvt_bf16<<<(16 * 32 + T - 1) / T, T, 0, stream>>>(W2_1, w2b1, 16 * 32);
    // relation projection (WMMA, async-LDS staged A, all relations per block)
    proj_gemm<<<NN / 16, 128, 0, stream>>>(ebf, wrt, proj);
    // attention + edge softmax
    att_edge<<<(EE * 32) / T, T, 0, stream>>>(proj, rel, src, dst, etype, att, segmax);
    ex_edge<<<(EE + T - 1) / T, T, 0, stream>>>(att, dst, segmax, ex, segsum);
    // layer 0
    agg64<<<(EE * 32) / T, T, 0, stream>>>(entity, ex, segsum, src, dst, Nh0);
    mlp_gemm64<<<NN / 16, 128, 0, stream>>>(entity, Nh0, w1b0, w2b0, b1_0, b2_0, z1, z2);
    norm32<<<(NN * 32) / T, T, 0, stream>>>(z1, z2, h1, out);
    ego_copy<<<(NN * 16 + T - 1) / T, T, 0, stream>>>(entity, out);
    // layer 1
    agg32<<<(EE * 32) / T, T, 0, stream>>>(h1, ex, segsum, src, dst, Nh1);
    mlp_gemm32<<<NN / 16, 64, 0, stream>>>(h1, Nh1, w1b1, w2b1, b1_1, b2_1, z1b, z2b);
    norm16<<<(NN * 32) / T, T, 0, stream>>>(z1b, z2b, out);
}
